// ComplexAttentionV3_68247030333573
// MI455X (gfx1250) — compile-verified, matrix operand order checked by reading, not run on hardware
//
#include <hip/hip_runtime.h>

// ---------------- problem constants ----------------
#define B_   2
#define T_   2048
#define D_   1024
#define H_   16
#define HD_  64
#define BT_  (B_ * T_)   // 4096 rows
#define K2_  (2 * D_)    // 2048 packed-K

typedef __attribute__((ext_vector_type(16))) __bf16 v16bf;
typedef __attribute__((ext_vector_type(8)))  float  v8f;

union FragU {
    v16bf  v;
    uint4  q[2];
    __bf16 b[16];
};

// Load one 16x32 bf16 WMMA fragment for a matrix whose K dimension is
// contiguous in memory (works for global or LDS pointers). CDNA5 16-bit
// A/B layout:
//   lanes 0-15 : halves 0-7 -> K = k0+0..7,   halves 8-15 -> K = k0+16..23
//   lanes 16-31: halves 0-7 -> K = k0+8..15,  halves 8-15 -> K = k0+24..31
__device__ __forceinline__ v16bf ld_rowK(const __bf16* rowbase, int k0, int lane) {
    int off = k0 + ((lane & 16) ? 8 : 0);
    FragU f;
    f.q[0] = *(const uint4*)(rowbase + off);
    f.q[1] = *(const uint4*)(rowbase + off + 16);
    return f.v;
}

__device__ __forceinline__ v8f wmma_bf16(v16bf a, v16bf b, v8f c) {
    return __builtin_amdgcn_wmma_f32_16x16x32_bf16(
        /*neg_a=*/false, a, /*neg_b=*/false, b,
        /*c_mod=*/(short)0, c, /*reuse_a=*/false, /*reuse_b=*/false);
}

// ---------------- pack [xr|xi] -> bf16 X[BT][2K] ----------------
__global__ void CA_cvt_x_kernel(const float* __restrict__ xr,
                                const float* __restrict__ xi,
                                __bf16* __restrict__ X) {
    int idx = blockIdx.x * 256 + threadIdx.x;       // BT_*K2_ threads
    int m = idx >> 11;
    int k = idx & (K2_ - 1);
    float v = (k < D_) ? xr[(size_t)m * D_ + k] : xi[(size_t)m * D_ + (k - D_)];
    X[idx] = (__bf16)v;
}

// Build transposed, sign-folded packed weights:
//   WTr[n][k] = k<D ?  wr[k][n] : -wi[k-D][n]   (real-output weights)
//   WTi[n][k] = k<D ?  wi[k][n] :  wr[k-D][n]   (imag-output weights)
__global__ void CA_cvt_w_kernel(const float* __restrict__ wr,
                                const float* __restrict__ wi,
                                __bf16* __restrict__ WTr,
                                __bf16* __restrict__ WTi) {
    int idx = blockIdx.x * 256 + threadIdx.x;       // D_*K2_ threads
    int n = idx >> 11;
    int k = idx & (K2_ - 1);
    if (k < D_) {
        WTr[idx] = (__bf16)wr[(size_t)k * D_ + n];
        WTi[idx] = (__bf16)wi[(size_t)k * D_ + n];
    } else {
        int kk = k - D_;
        WTr[idx] = (__bf16)(-wi[(size_t)kk * D_ + n]);
        WTi[idx] = (__bf16)wr[(size_t)kk * D_ + n];
    }
}

// ---------------- WMMA GEMM: C[M][N] = A[M][K] * Bt[N][K]^T ----------------
// Each wave computes a 32x64 output block (2 A-frags x 4 B-frags -> 8 WMMAs
// per 32-deep K chunk). Workgroup: 8 waves -> 64 (M) x 256 (N).
// mode 0: store bf16 row-major C[m*N+n]
// mode 1: store bf16 transposed  C[n*M+m]   (for V^T)
// mode 2: store fp32 row-major   C[m*N+n]   (final outputs)
__launch_bounds__(256)
__global__ void CA_gemm_wmma_kernel(const __bf16* __restrict__ A,
                                    const __bf16* __restrict__ Bt,
                                    void* __restrict__ Cout,
                                    int M, int N, int K, int mode) {
    const int lane = threadIdx.x & 31;
    const int wave = threadIdx.x >> 5;   // 0..7
    const int wm = wave >> 2;            // 0..1
    const int wn = wave & 3;             // 0..3
    const int m0 = blockIdx.y * 64 + wm * 32;
    const int n0 = blockIdx.x * 256 + wn * 64;
    const int l15 = lane & 15;
    const int hi  = (lane >> 4) * 8;

    v8f acc[2][4];
    for (int mi = 0; mi < 2; ++mi)
        for (int ni = 0; ni < 4; ++ni)
            acc[mi][ni] = (v8f){};

    const __bf16* ar0 = A  + (size_t)(m0 + l15)      * K;
    const __bf16* ar1 = A  + (size_t)(m0 + 16 + l15) * K;
    const __bf16* br0 = Bt + (size_t)(n0 + l15)      * K;
    const __bf16* br1 = Bt + (size_t)(n0 + 16 + l15) * K;
    const __bf16* br2 = Bt + (size_t)(n0 + 32 + l15) * K;
    const __bf16* br3 = Bt + (size_t)(n0 + 48 + l15) * K;

    for (int k0 = 0; k0 < K; k0 += 32) {
        v16bf a0 = ld_rowK(ar0, k0, lane);
        v16bf a1 = ld_rowK(ar1, k0, lane);
        v16bf b0 = ld_rowK(br0, k0, lane);
        v16bf b1 = ld_rowK(br1, k0, lane);
        v16bf b2 = ld_rowK(br2, k0, lane);
        v16bf b3 = ld_rowK(br3, k0, lane);
        acc[0][0] = wmma_bf16(a0, b0, acc[0][0]);
        acc[0][1] = wmma_bf16(a0, b1, acc[0][1]);
        acc[0][2] = wmma_bf16(a0, b2, acc[0][2]);
        acc[0][3] = wmma_bf16(a0, b3, acc[0][3]);
        acc[1][0] = wmma_bf16(a1, b0, acc[1][0]);
        acc[1][1] = wmma_bf16(a1, b1, acc[1][1]);
        acc[1][2] = wmma_bf16(a1, b2, acc[1][2]);
        acc[1][3] = wmma_bf16(a1, b3, acc[1][3]);
    }

    for (int mi = 0; mi < 2; ++mi) {
        for (int ni = 0; ni < 4; ++ni) {
            int n = n0 + ni * 16 + l15;
            for (int r = 0; r < 8; ++r) {
                int m = m0 + mi * 16 + hi + r;   // C/D layout: lane->N, vgpr->M (+8 hi half)
                float v = acc[mi][ni][r];
                if (mode == 0)      ((__bf16*)Cout)[(size_t)m * N + n] = (__bf16)v;
                else if (mode == 1) ((__bf16*)Cout)[(size_t)n * M + m] = (__bf16)v;
                else                ((float*)Cout)[(size_t)m * N + n]  = v;
            }
        }
    }
}

// ---------------- RoPE (elementwise complex rotation) ----------------
__global__ void CA_rope_kernel(__bf16* __restrict__ Xr, __bf16* __restrict__ Xi) {
    int idx = blockIdx.x * 256 + threadIdx.x;   // BT_*D_ threads
    int m = idx >> 10;
    int c = idx & (D_ - 1);
    int t = m & (T_ - 1);
    int d = c & (HD_ - 1);
    int j = d & 31;                              // concat([freqs,freqs])
    float ang = (float)t * powf(10000.0f, -(float)j / 32.0f);
    float sv = sinf(ang), cv = cosf(ang);
    float xr = (float)Xr[idx], xi = (float)Xi[idx];
    Xr[idx] = (__bf16)(xr * cv - xi * sv);
    Xi[idx] = (__bf16)(xr * sv + xi * cv);
}

// ---------------- attention: 16 query rows per WG, 4 waves ----------------
__launch_bounds__(128)
__global__ void CA_attn_wmma_kernel(const __bf16* __restrict__ Qr, const __bf16* __restrict__ Qi,
                                    const __bf16* __restrict__ Kr, const __bf16* __restrict__ Ki,
                                    const __bf16* __restrict__ VrT, const __bf16* __restrict__ ViT,
                                    __bf16* __restrict__ AO) {
    __shared__ float  S[16][T_];    // 128 KB raw scores / exp values
    __shared__ __bf16 P[16][T_];    //  64 KB normalized probabilities (bf16)
    __shared__ float  red[16][8];

    const int lane = threadIdx.x & 31;
    const int wave = threadIdx.x >> 5;   // 0..3
    const int l15 = lane & 15;
    const int hi  = (lane >> 4) * 8;
    const int bh = blockIdx.y;
    const int b  = bh >> 4;
    const int h  = bh & 15;
    const int row0 = b * T_ + blockIdx.x * 16;

    // ---- phase 1: S = (qr kr^T + qi ki^T) * scale ----
    const __bf16* qrb = Qr + (size_t)(row0 + l15) * D_ + h * HD_;
    const __bf16* qib = Qi + (size_t)(row0 + l15) * D_ + h * HD_;
    v16bf qr0 = ld_rowK(qrb, 0, lane), qr1 = ld_rowK(qrb, 32, lane);
    v16bf qi0 = ld_rowK(qib, 0, lane), qi1 = ld_rowK(qib, 32, lane);

    for (int kb = 0; kb < 32; ++kb) {
        int j0 = wave * 512 + kb * 16;          // waves own disjoint key ranges
        const __bf16* krb = Kr + (size_t)(b * T_ + j0 + l15) * D_ + h * HD_;
        const __bf16* kib = Ki + (size_t)(b * T_ + j0 + l15) * D_ + h * HD_;
        v16bf kr0 = ld_rowK(krb, 0, lane), kr1 = ld_rowK(krb, 32, lane);
        v16bf ki0 = ld_rowK(kib, 0, lane), ki1 = ld_rowK(kib, 32, lane);
        v8f acc = {};
        acc = wmma_bf16(qr0, kr0, acc);
        acc = wmma_bf16(qr1, kr1, acc);
        acc = wmma_bf16(qi0, ki0, acc);
        acc = wmma_bf16(qi1, ki1, acc);
        int j = j0 + l15;
        for (int r = 0; r < 8; ++r) S[hi + r][j] = acc[r] * 0.125f;  // 1/sqrt(64)
    }
    __syncthreads();

    // ---- phase 2: softmax over each of the 16 rows (8 threads/row),
    //      normalized result written once as bf16 into P ----
    {
        int row = threadIdx.x >> 3;
        int seg = threadIdx.x & 7;
        int c0 = seg * 256;
        float lmax = -1e30f;
        for (int c = 0; c < 256; ++c) lmax = fmaxf(lmax, S[row][c0 + c]);
        red[row][seg] = lmax;
        __syncthreads();
        float rmax = red[row][0];
        for (int s2 = 1; s2 < 8; ++s2) rmax = fmaxf(rmax, red[row][s2]);
        __syncthreads();
        float lsum = 0.f;
        for (int c = 0; c < 256; ++c) {
            float e = __expf(S[row][c0 + c] - rmax);
            S[row][c0 + c] = e;
            lsum += e;
        }
        red[row][seg] = lsum;
        __syncthreads();
        float rsum = 0.f;
        for (int s2 = 0; s2 < 8; ++s2) rsum += red[row][s2];
        float rinv = 1.0f / rsum;
        for (int c = 0; c < 256; ++c)
            P[row][c0 + c] = (__bf16)(S[row][c0 + c] * rinv);
    }
    __syncthreads();

    // ---- phase 3: O = P @ V (each wave owns 16 head-dim columns);
    //      A-fragments come straight from bf16 LDS via b128 loads ----
    int nb = wave * 16;
    const __bf16* prow = &P[l15][0];
    const __bf16* vrb = VrT + (size_t)(h * HD_ + nb + l15) * BT_ + b * T_;
    const __bf16* vib = ViT + (size_t)(h * HD_ + nb + l15) * BT_ + b * T_;
    v8f o_r = {}, o_i = {};
    for (int k0 = 0; k0 < T_; k0 += 32) {
        v16bf pf = ld_rowK(prow, k0, lane);
        v16bf br = ld_rowK(vrb, k0, lane);
        v16bf bi = ld_rowK(vib, k0, lane);
        o_r = wmma_bf16(pf, br, o_r);
        o_i = wmma_bf16(pf, bi, o_i);
    }
    int n = nb + l15;
    for (int r = 0; r < 8; ++r) {
        int grow = row0 + hi + r;
        AO[(size_t)grow * K2_ + h * HD_ + n]      = (__bf16)o_r[r];
        AO[(size_t)grow * K2_ + D_ + h * HD_ + n] = (__bf16)o_i[r];
    }
}

// ---------------- host launcher ----------------
extern "C" void kernel_launch(void* const* d_in, const int* in_sizes, int n_in,
                              void* d_out, int out_size, void* d_ws, size_t ws_size,
                              hipStream_t stream) {
    const float* xr   = (const float*)d_in[0];
    const float* xi   = (const float*)d_in[1];
    const float* q_wr = (const float*)d_in[2];
    const float* q_wi = (const float*)d_in[3];
    const float* k_wr = (const float*)d_in[4];
    const float* k_wi = (const float*)d_in[5];
    const float* v_wr = (const float*)d_in[6];
    const float* v_wi = (const float*)d_in[7];
    const float* o_wr = (const float*)d_in[8];
    const float* o_wi = (const float*)d_in[9];

    char* ws = (char*)d_ws;
    size_t off = 0;
    auto nxt = [&](size_t bytes) -> char* {
        char* p = ws + off;
        off += (bytes + 255) & ~(size_t)255;
        return p;
    };

    __bf16* Xbf  = (__bf16*)nxt((size_t)BT_ * K2_ * 2);
    __bf16* WqrT = (__bf16*)nxt((size_t)D_ * K2_ * 2);
    __bf16* WqiT = (__bf16*)nxt((size_t)D_ * K2_ * 2);
    __bf16* WkrT = (__bf16*)nxt((size_t)D_ * K2_ * 2);
    __bf16* WkiT = (__bf16*)nxt((size_t)D_ * K2_ * 2);
    __bf16* WvrT = (__bf16*)nxt((size_t)D_ * K2_ * 2);
    __bf16* WviT = (__bf16*)nxt((size_t)D_ * K2_ * 2);
    __bf16* WorT = (__bf16*)nxt((size_t)D_ * K2_ * 2);
    __bf16* WoiT = (__bf16*)nxt((size_t)D_ * K2_ * 2);
    __bf16* Qr   = (__bf16*)nxt((size_t)BT_ * D_ * 2);
    __bf16* Qi   = (__bf16*)nxt((size_t)BT_ * D_ * 2);
    __bf16* Krb  = (__bf16*)nxt((size_t)BT_ * D_ * 2);
    __bf16* Kib  = (__bf16*)nxt((size_t)BT_ * D_ * 2);
    __bf16* VrT  = (__bf16*)nxt((size_t)D_ * BT_ * 2);
    __bf16* ViT  = (__bf16*)nxt((size_t)D_ * BT_ * 2);
    __bf16* AO   = (__bf16*)nxt((size_t)BT_ * K2_ * 2);

    // pack inputs / weights to bf16
    CA_cvt_x_kernel<<<(BT_ * K2_) / 256, 256, 0, stream>>>(xr, xi, Xbf);
    CA_cvt_w_kernel<<<(D_ * K2_) / 256, 256, 0, stream>>>(q_wr, q_wi, WqrT, WqiT);
    CA_cvt_w_kernel<<<(D_ * K2_) / 256, 256, 0, stream>>>(k_wr, k_wi, WkrT, WkiT);
    CA_cvt_w_kernel<<<(D_ * K2_) / 256, 256, 0, stream>>>(v_wr, v_wi, WvrT, WviT);
    CA_cvt_w_kernel<<<(D_ * K2_) / 256, 256, 0, stream>>>(o_wr, o_wi, WorT, WoiT);

    dim3 gg(D_ / 256, BT_ / 64);   // (4, 64)
    // projections (WMMA)
    CA_gemm_wmma_kernel<<<gg, 256, 0, stream>>>(Xbf, WqrT, Qr,  BT_, D_, K2_, 0);
    CA_gemm_wmma_kernel<<<gg, 256, 0, stream>>>(Xbf, WqiT, Qi,  BT_, D_, K2_, 0);
    CA_gemm_wmma_kernel<<<gg, 256, 0, stream>>>(Xbf, WkrT, Krb, BT_, D_, K2_, 0);
    CA_gemm_wmma_kernel<<<gg, 256, 0, stream>>>(Xbf, WkiT, Kib, BT_, D_, K2_, 0);
    CA_gemm_wmma_kernel<<<gg, 256, 0, stream>>>(Xbf, WvrT, VrT, BT_, D_, K2_, 1); // transposed store
    CA_gemm_wmma_kernel<<<gg, 256, 0, stream>>>(Xbf, WviT, ViT, BT_, D_, K2_, 1);

    // RoPE on Q and K
    CA_rope_kernel<<<(BT_ * D_) / 256, 256, 0, stream>>>(Qr, Qi);
    CA_rope_kernel<<<(BT_ * D_) / 256, 256, 0, stream>>>(Krb, Kib);

    // attention
    CA_attn_wmma_kernel<<<dim3(T_ / 16, B_ * H_), 128, 0, stream>>>(
        Qr, Qi, Krb, Kib, VrT, ViT, AO);

    // final complex output projection -> fp32 d_out (final_r then final_i)
    float* outr = (float*)d_out;
    float* outi = outr + (size_t)BT_ * D_;
    CA_gemm_wmma_kernel<<<gg, 256, 0, stream>>>(AO, WorT, outr, BT_, D_, K2_, 2);
    CA_gemm_wmma_kernel<<<gg, 256, 0, stream>>>(AO, WoiT, outi, BT_, D_, K2_, 2);
}
